// Discriminator_32933809225818
// MI455X (gfx1250) — matrix operand order repaired and checked by reading, not compiled, not run on hardware
//
#include <hip/hip_runtime.h>
#include <math.h>

typedef __attribute__((ext_vector_type(16))) _Float16 v16h;
typedef __attribute__((ext_vector_type(8)))  float    v8f;
typedef int v4i __attribute__((vector_size(16)));

#define NN   2048
#define BB   4
#define TT   8
#define DHH  64
#define FF   128
#define ROWS (BB*NN)   /* 8192 */

#if defined(__has_builtin)
#if __has_builtin(__builtin_amdgcn_global_load_async_to_lds_b128) && __has_builtin(__builtin_amdgcn_s_wait_asynccnt)
#define USE_ASYNC_LDS 1
#endif
#endif

// ---------------------------------------------------------------- prep ----

__global__ void k_convert_A(const float* __restrict__ A, _Float16* __restrict__ Ah, int n) {
    int i = blockIdx.x * 256 + threadIdx.x;
    if (i < n) Ah[i] = (_Float16)A[i];
}

// power iteration: sigma = top singular value of W [R x C], R<=384, C<=128
__global__ void k_sigma(const float* __restrict__ W, int R, int C, float* __restrict__ sig) {
    __shared__ float v[128];
    __shared__ float u[384];
    __shared__ float red[128];
    int tid = threadIdx.x;  // 128 threads
    if (tid < C) v[tid] = 1.0f;
    __syncthreads();
    float lambda = 0.0f;
    for (int it = 0; it < 64; ++it) {
        for (int r = tid; r < R; r += 128) {
            float s = 0.0f;
            for (int c = 0; c < C; ++c) s += W[r * C + c] * v[c];
            u[r] = s;
        }
        __syncthreads();
        float w = 0.0f;
        if (tid < C) {
            for (int r = 0; r < R; ++r) w += W[r * C + tid] * u[r];
        }
        red[tid] = (tid < C) ? w * w : 0.0f;
        __syncthreads();
        for (int s = 64; s > 0; s >>= 1) {
            if (tid < s) red[tid] += red[tid + s];
            __syncthreads();
        }
        lambda = sqrtf(red[0]);   // ~ sigma^2
        float inv = (lambda > 0.0f) ? 1.0f / lambda : 0.0f;
        if (tid < C) v[tid] = w * inv;
        __syncthreads();
    }
    if (tid == 0) sig[0] = sqrtf(lambda);
}

// W [K][IN][OUT] fp32 -> Wcm [K][OUT][IN] f16, scaled by 1/sigma
__global__ void k_scale_w(const float* __restrict__ W, const float* __restrict__ sig,
                          _Float16* __restrict__ Wcm, int K_, int IN, int OUT) {
    int idx = blockIdx.x * 256 + threadIdx.x;
    int tot = K_ * IN * OUT;
    if (idx >= tot) return;
    float inv = 1.0f / sig[0];
    int k   = idx / (IN * OUT);
    int rem = idx % (IN * OUT);
    int o   = rem / IN;
    int i   = rem % IN;
    Wcm[idx] = (_Float16)(W[(k * IN + i) * OUT + o] * inv);
}

__global__ void k_zero_f32(float* __restrict__ p, int n) {
    int i = blockIdx.x * 256 + threadIdx.x;
    if (i < n) p[i] = 0.0f;
}

// -------------------------------------------------------- elementwise ----

// XH[row][0:64]=x_t, [64:128]=h    row = b*NN+n
__global__ void k_build_xh(const float* __restrict__ X, const float* __restrict__ Hseq,
                           const float* __restrict__ Hf, _Float16* __restrict__ XH,
                           int t, int isblk0) {
    int gid = blockIdx.x * 256 + threadIdx.x;
    if (gid >= ROWS * FF) return;
    int row = gid >> 7, f = gid & 127;
    int b = row >> 11, n = row & (NN - 1);
    float val;
    if (f < 64) {
        val = isblk0 ? X[(((size_t)b * TT + t) * NN + n) * 64 + f]
                     : Hseq[(((size_t)t * BB + b) * NN + n) * 64 + f];
    } else {
        val = Hf[(size_t)row * 64 + (f - 64)];
    }
    XH[gid] = (_Float16)val;
}

// XH2[row][0:64]=x_t, [64:128]=r*h with r = G[row][0:64]
__global__ void k_build_xh2(const float* __restrict__ X, const float* __restrict__ Hseq,
                            const float* __restrict__ G, const float* __restrict__ Hf,
                            _Float16* __restrict__ XH2, int t, int isblk0) {
    int gid = blockIdx.x * 256 + threadIdx.x;
    if (gid >= ROWS * FF) return;
    int row = gid >> 7, f = gid & 127;
    int b = row >> 11, n = row & (NN - 1);
    float val;
    if (f < 64) {
        val = isblk0 ? X[(((size_t)b * TT + t) * NN + n) * 64 + f]
                     : Hseq[(((size_t)t * BB + b) * NN + n) * 64 + f];
    } else {
        int j = f - 64;
        val = G[(size_t)row * FF + j] * Hf[(size_t)row * 64 + j];
    }
    XH2[gid] = (_Float16)val;
}

// ----------------------------------------------------------- A @ S -------
// Sout[b,n,f] = sum_m Ah[n,m]*Sin[b,m,f]; grid (NN/64, FF/64, BB), block 128
__global__ void __launch_bounds__(128)
k_gemm_AS(const _Float16* __restrict__ Ah, const _Float16* __restrict__ Sin,
          _Float16* __restrict__ Sout) {
    __shared__ __align__(16) _Float16 As[64 * 40];  // row-major, pitch 40
    __shared__ __align__(16) _Float16 Ss[64 * 36];  // col-major [col][k], pitch 36

    const int rowBase = blockIdx.x * 64;
    const int colBase = blockIdx.y * 64;
    const _Float16* S = Sin  + (size_t)blockIdx.z * NN * FF;
    _Float16*       O = Sout + (size_t)blockIdx.z * NN * FF;

    const int tid  = threadIdx.x;
    const int wave = tid >> 5;
    const int lane = tid & 31;
    const int half = lane >> 4;
    const int l15  = lane & 15;

    v8f acc[4] = {};

    for (int k0 = 0; k0 < NN; k0 += 32) {
        // stage A tile 64x32 (16B chunks, 2 per thread)
#pragma unroll
        for (int c = 0; c < 2; ++c) {
            int chunk = tid + c * 128;        // 0..255
            int m  = chunk >> 2;              // row in tile
            int ko = (chunk & 3) * 8;         // k offset
            const _Float16* gp = Ah + (size_t)(rowBase + m) * NN + k0 + ko;
            _Float16* lp = &As[m * 40 + ko];
#if defined(USE_ASYNC_LDS)
            __builtin_amdgcn_global_load_async_to_lds_b128(
                (__attribute__((address_space(1))) v4i*)gp,
                (__attribute__((address_space(3))) v4i*)lp, 0, 0);
#else
            *(uint4*)lp = *(const uint4*)gp;
#endif
        }
        // stage S tile 32x64, transposed into Ss[col][k]
#pragma unroll
        for (int c = 0; c < 2; ++c) {
            int chunk = tid + c * 128;        // 0..255
            int kk = chunk >> 3;              // 0..31
            int f0 = (chunk & 7) * 8;
            const _Float16* gp = S + (size_t)(k0 + kk) * FF + colBase + f0;
            uint4 d = *(const uint4*)gp;
            const _Float16* hp = (const _Float16*)&d;
#pragma unroll
            for (int i = 0; i < 8; ++i) Ss[(f0 + i) * 36 + kk] = hp[i];
        }
#if defined(USE_ASYNC_LDS)
        __builtin_amdgcn_s_wait_asynccnt(0);
#endif
        __syncthreads();

        // A fragment for this wave's 16-row strip
        v16h a;
#pragma unroll
        for (int j = 0; j < 8; ++j) {
            int kk = 16 * half + 2 * j;
            const _Float16* p = &As[(wave * 16 + l15) * 40 + kk];
            a[2 * j] = p[0]; a[2 * j + 1] = p[1];
        }
#pragma unroll
        for (int ct = 0; ct < 4; ++ct) {
            v16h bf;
#pragma unroll
            for (int j = 0; j < 8; ++j) {
                int kb = 16 * half + 2 * j;
                const _Float16* p = &Ss[(ct * 16 + l15) * 36 + kb];
                bf[2 * j] = p[0]; bf[2 * j + 1] = p[1];
            }
            acc[ct] = __builtin_amdgcn_wmma_f32_16x16x32_f16(
                false, a, false, bf, (short)0, acc[ct], false, false);
        }
        __syncthreads();
    }

#pragma unroll
    for (int ct = 0; ct < 4; ++ct) {
#pragma unroll
        for (int r = 0; r < 8; ++r) {
            int gm = rowBase + wave * 16 + r + 8 * half;
            int gn = colBase + ct * 16 + l15;
            O[(size_t)gm * FF + gn] = (_Float16)acc[ct][r];
        }
    }
}

// ---------------------------------------------- gate: sigmoid(3-term) ----
// G[row][o] = sigmoid( XH@W0 + S1@W1 + S2@W2 + bg ), o in [0,128)
__global__ void __launch_bounds__(256)
k_gemm_gate(const _Float16* __restrict__ XH, const _Float16* __restrict__ S1,
            const _Float16* __restrict__ S2, const _Float16* __restrict__ Wcm,
            const float* __restrict__ Bg, float* __restrict__ G) {
    const int tid  = threadIdx.x;
    const int wave = tid >> 5;
    const int lane = tid & 31;
    const int half = lane >> 4;
    const int l15  = lane & 15;
    const int mw = wave >> 1, nw = wave & 1;
    const int rowBase = blockIdx.x * 64 + mw * 16;
    const int colBase = nw * 64;

    const _Float16* Xs[3] = {XH, S1, S2};
    v8f acc[4] = {};

    for (int term = 0; term < 3; ++term) {
        const _Float16* Xp = Xs[term];
        const _Float16* Wp = Wcm + (size_t)term * FF * FF;  // [o][i] col-major
#pragma unroll
        for (int ks = 0; ks < 4; ++ks) {
            int k0 = ks * 32;
            v16h a;
#pragma unroll
            for (int j = 0; j < 8; ++j) {
                int kk = 16 * half + 2 * j;
                const _Float16* p = Xp + (size_t)(rowBase + l15) * FF + k0 + kk;
                a[2 * j] = p[0]; a[2 * j + 1] = p[1];
            }
#pragma unroll
            for (int ct = 0; ct < 4; ++ct) {
                int o = colBase + ct * 16 + l15;
                v16h bf;
#pragma unroll
                for (int j = 0; j < 8; ++j) {
                    int kb = 16 * half + 2 * j;
                    const _Float16* p = Wp + (size_t)o * FF + k0 + kb;
                    bf[2 * j] = p[0]; bf[2 * j + 1] = p[1];
                }
                acc[ct] = __builtin_amdgcn_wmma_f32_16x16x32_f16(
                    false, a, false, bf, (short)0, acc[ct], false, false);
            }
        }
    }
#pragma unroll
    for (int ct = 0; ct < 4; ++ct) {
#pragma unroll
        for (int r = 0; r < 8; ++r) {
            int gm = rowBase + r + 8 * half;
            int go = colBase + ct * 16 + l15;
            float v = acc[ct][r] + Bg[go];
            G[(size_t)gm * FF + go] = 1.0f / (1.0f + expf(-v));
        }
    }
}

// ------------------------------- candidate + GRU update (fused epilog) ---
__global__ void __launch_bounds__(256)
k_gemm_cand_update(const _Float16* __restrict__ XH2, const _Float16* __restrict__ S1,
                   const _Float16* __restrict__ S2, const _Float16* __restrict__ Wcm,
                   const float* __restrict__ Bc, const float* __restrict__ G,
                   float* __restrict__ Hf, float* __restrict__ HseqT) {
    const int tid  = threadIdx.x;
    const int wave = tid >> 5;
    const int lane = tid & 31;
    const int half = lane >> 4;
    const int l15  = lane & 15;
    const int mw = wave >> 1, nw = wave & 1;
    const int rowBase = blockIdx.x * 64 + mw * 16;
    const int colBase = nw * 32;

    const _Float16* Xs[3] = {XH2, S1, S2};
    v8f acc[2] = {};

    for (int term = 0; term < 3; ++term) {
        const _Float16* Xp = Xs[term];
        const _Float16* Wp = Wcm + (size_t)term * DHH * FF;  // [o:64][i:128] col-major
#pragma unroll
        for (int ks = 0; ks < 4; ++ks) {
            int k0 = ks * 32;
            v16h a;
#pragma unroll
            for (int j = 0; j < 8; ++j) {
                int kk = 16 * half + 2 * j;
                const _Float16* p = Xp + (size_t)(rowBase + l15) * FF + k0 + kk;
                a[2 * j] = p[0]; a[2 * j + 1] = p[1];
            }
#pragma unroll
            for (int ct = 0; ct < 2; ++ct) {
                int o = colBase + ct * 16 + l15;
                v16h bf;
#pragma unroll
                for (int j = 0; j < 8; ++j) {
                    int kb = 16 * half + 2 * j;
                    const _Float16* p = Wp + (size_t)o * FF + k0 + kb;
                    bf[2 * j] = p[0]; bf[2 * j + 1] = p[1];
                }
                acc[ct] = __builtin_amdgcn_wmma_f32_16x16x32_f16(
                    false, a, false, bf, (short)0, acc[ct], false, false);
            }
        }
    }
#pragma unroll
    for (int ct = 0; ct < 2; ++ct) {
#pragma unroll
        for (int r = 0; r < 8; ++r) {
            int gm = rowBase + r + 8 * half;
            int go = colBase + ct * 16 + l15;
            float c = tanhf(acc[ct][r] + Bc[go]);
            float u = G[(size_t)gm * FF + 64 + go];
            float h = Hf[(size_t)gm * DHH + go];
            float hn = u * h + (1.0f - u) * c;
            Hf[(size_t)gm * DHH + go] = hn;
            if (HseqT) HseqT[(size_t)gm * DHH + go] = hn;
        }
    }
}

// ----------------------------------------------------------- output ------

__global__ void k_out_partial(const float* __restrict__ Hf, const float* __restrict__ Wout,
                              float* __restrict__ part) {
    __shared__ float red[256];
    int tid = threadIdx.x;
    float s = 0.0f;
    for (int i = blockIdx.x * 256 + tid; i < ROWS * DHH; i += 64 * 256)
        s += Hf[i] * Wout[i & 63];
    red[tid] = s;
    __syncthreads();
    for (int d = 128; d > 0; d >>= 1) {
        if (tid < d) red[tid] += red[tid + d];
        __syncthreads();
    }
    if (tid == 0) part[blockIdx.x] = red[0];
}

__global__ void k_out_final(const float* __restrict__ part, const float* __restrict__ Wout,
                            const float* __restrict__ bout, float* __restrict__ out) {
    __shared__ float r1[64], r2[64];
    int tid = threadIdx.x;  // 64
    r1[tid] = part[tid];
    float w = Wout[tid];
    r2[tid] = w * w;
    __syncthreads();
    for (int d = 32; d > 0; d >>= 1) {
        if (tid < d) { r1[tid] += r1[tid + d]; r2[tid] += r2[tid + d]; }
        __syncthreads();
    }
    if (tid == 0) out[0] = r1[0] / ((float)ROWS * sqrtf(r2[0])) + bout[0];
}

// ------------------------------------------------------------- host ------

extern "C" void kernel_launch(void* const* d_in, const int* in_sizes, int n_in,
                              void* d_out, int out_size, void* d_ws, size_t ws_size,
                              hipStream_t stream) {
    (void)in_sizes; (void)n_in; (void)out_size; (void)ws_size;
    const float* X    = (const float*)d_in[0];
    const float* A    = (const float*)d_in[2];
    const float* Wg   = (const float*)d_in[4];
    const float* bg   = (const float*)d_in[5];
    const float* Wc   = (const float*)d_in[6];
    const float* bc   = (const float*)d_in[7];
    const float* Wout = (const float*)d_in[8];
    const float* bout = (const float*)d_in[9];

    char* w = (char*)d_ws;
    size_t off = 0;
    auto take = [&](size_t bytes) -> void* {
        void* p = w + off;
        off += (bytes + 255) & ~(size_t)255;
        return p;
    };
    _Float16* Ah   = (_Float16*)take((size_t)NN * NN * 2);
    _Float16* WgCM = (_Float16*)take((size_t)2 * 3 * 128 * 128 * 2);
    _Float16* WcCM = (_Float16*)take((size_t)2 * 3 * 64 * 128 * 2);
    float* sig     = (float*)take(16);
    float* Hf      = (float*)take((size_t)ROWS * DHH * 4);
    float* Hseq    = (float*)take((size_t)TT * ROWS * DHH * 4);
    _Float16* XH   = (_Float16*)take((size_t)ROWS * FF * 2);
    _Float16* S1   = (_Float16*)take((size_t)ROWS * FF * 2);
    _Float16* S2   = (_Float16*)take((size_t)ROWS * FF * 2);
    float* G       = (float*)take((size_t)ROWS * FF * 4);
    float* part    = (float*)take(64 * 4);

    // ---- prep: A to f16, spectral-normalized col-major f16 weights ----
    k_convert_A<<<(NN * NN + 255) / 256, 256, 0, stream>>>(A, Ah, NN * NN);
    for (int blk = 0; blk < 2; ++blk) {
        k_sigma<<<1, 128, 0, stream>>>(Wg + (size_t)blk * 3 * 128 * 128, 384, 128, sig + blk);
        k_sigma<<<1, 128, 0, stream>>>(Wc + (size_t)blk * 3 * 128 * 64, 384, 64, sig + 2 + blk);
        k_scale_w<<<(3 * 128 * 128 + 255) / 256, 256, 0, stream>>>(
            Wg + (size_t)blk * 3 * 128 * 128, sig + blk,
            WgCM + (size_t)blk * 3 * 128 * 128, 3, 128, 128);
        k_scale_w<<<(3 * 128 * 64 + 255) / 256, 256, 0, stream>>>(
            Wc + (size_t)blk * 3 * 128 * 64, sig + 2 + blk,
            WcCM + (size_t)blk * 3 * 64 * 128, 3, 128, 64);
    }

    dim3 gAS(NN / 64, FF / 64, BB);
    const int ewBlocks = (ROWS * FF + 255) / 256;

    for (int blk = 0; blk < 2; ++blk) {
        const int isblk0 = (blk == 0) ? 1 : 0;
        const _Float16* WgB = WgCM + (size_t)blk * 3 * 128 * 128;
        const _Float16* WcB = WcCM + (size_t)blk * 3 * 64 * 128;
        const float* bgB = bg + (size_t)blk * 128;
        const float* bcB = bc + (size_t)blk * 64;

        k_zero_f32<<<(ROWS * DHH + 255) / 256, 256, 0, stream>>>(Hf, ROWS * DHH);

        for (int t = 0; t < TT; ++t) {
            k_build_xh<<<ewBlocks, 256, 0, stream>>>(X, Hseq, Hf, XH, t, isblk0);
            k_gemm_AS<<<gAS, 128, 0, stream>>>(Ah, XH, S1);
            k_gemm_AS<<<gAS, 128, 0, stream>>>(Ah, S1, S2);
            k_gemm_gate<<<ROWS / 64, 256, 0, stream>>>(XH, S1, S2, WgB, bgB, G);
            k_build_xh2<<<ewBlocks, 256, 0, stream>>>(X, Hseq, G, Hf, XH, t, isblk0);
            k_gemm_AS<<<gAS, 128, 0, stream>>>(Ah, XH, S1);
            k_gemm_AS<<<gAS, 128, 0, stream>>>(Ah, S1, S2);
            float* HseqT = (blk == 0) ? (Hseq + (size_t)t * ROWS * DHH) : nullptr;
            k_gemm_cand_update<<<ROWS / 64, 256, 0, stream>>>(XH, S1, S2, WcB, bcB, G, Hf, HseqT);
        }
    }

    k_out_partial<<<64, 256, 0, stream>>>(Hf, Wout, part);
    k_out_final<<<1, 64, 0, stream>>>(part, Wout, bout, (float*)d_out);
}
